// SelfAttn_5849745457258
// MI455X (gfx1250) — compile-verified, matrix operand order checked by reading, not run on hardware
//
#include <hip/hip_runtime.h>
#include <hip/hip_bf16.h>

// ---------------------------------------------------------------------------
// Problem constants
#define BATCH 4
#define SEQ   2048
#define EMB   768
#define NHEAD 12
#define HDIM  64
#define MROWS (BATCH * SEQ)          // 8192
#define NQKV  (3 * EMB)              // 2304
#define LN_EPS 1e-5f

typedef __attribute__((ext_vector_type(16))) _Float16 v16h;
typedef __attribute__((ext_vector_type(8)))  _Float16 v8h;
typedef __attribute__((ext_vector_type(8)))  float    v8f;

union V16U { v16h v; v8h h[2]; _Float16 e[16]; };
union V8HU { v8h v; _Float16 e[8]; };

// ---------------------------------------------------------------------------
// WMMA helpers

__device__ inline v8f wmma_f16(v16h a, v16h b, v8f c) {
  // D = A(16x32 f16) x B(32x16 f16) + C(16x16 f32)
  return __builtin_amdgcn_wmma_f32_16x16x32_f16(
      /*neg_a=*/false, a, /*neg_b=*/false, b,
      /*c_mod=*/(short)0, c, /*reuse_a=*/false, /*reuse_b=*/false);
}

// Load a 16x32 f16 fragment (A-layout; identical pattern serves column-major B).
// Per ISA 7.12.2: lane l -> row (l&15); hi=l>>4 selects K-halves:
//   VGPR0..3 hold K = hi*8 .. hi*8+7    (one 16B chunk)
//   VGPR4..7 hold K = 16+hi*8 .. 23+hi*8 (second 16B chunk)
__device__ inline v16h load_k32(const _Float16* rowbase, int ld, int k0) {
  const int lane = threadIdx.x & 31;
  const int r = lane & 15;
  const int hi = lane >> 4;
  const _Float16* p = rowbase + (size_t)r * (size_t)ld + (size_t)(k0 + hi * 8);
  V16U u;
  u.h[0] = *(const v8h*)(p);
  u.h[1] = *(const v8h*)(p + 16);
  return u.v;
}

// ---------------------------------------------------------------------------
// 1) LayerNorm -> f16 activations
__global__ __launch_bounds__(256) void ln_kernel(const float* __restrict__ feat,
                                                 const float* __restrict__ g,
                                                 const float* __restrict__ b,
                                                 _Float16* __restrict__ xln) {
  const int row = blockIdx.x;                    // 0..MROWS-1
  const float* x = feat + (size_t)row * EMB;
  float v[3], s = 0.f, s2 = 0.f;
#pragma unroll
  for (int i = 0; i < 3; ++i) {
    float t = x[threadIdx.x + 256 * i];
    v[i] = t; s += t; s2 += t * t;
  }
#pragma unroll
  for (int m = 1; m < 32; m <<= 1) { s += __shfl_xor(s, m, 32); s2 += __shfl_xor(s2, m, 32); }
  __shared__ float rs[8], rs2[8];
  const int wid = threadIdx.x >> 5, lane = threadIdx.x & 31;
  if (lane == 0) { rs[wid] = s; rs2[wid] = s2; }
  __syncthreads();
  float S = 0.f, S2 = 0.f;
#pragma unroll
  for (int i = 0; i < 8; ++i) { S += rs[i]; S2 += rs2[i]; }
  const float mu   = S * (1.0f / EMB);
  const float var  = S2 * (1.0f / EMB) - mu * mu;
  const float rstd = rsqrtf(var + LN_EPS);
#pragma unroll
  for (int i = 0; i < 3; ++i) {
    const int idx = threadIdx.x + 256 * i;
    xln[(size_t)row * EMB + idx] = (_Float16)((v[i] - mu) * rstd * g[idx] + b[idx]);
  }
}

// ---------------------------------------------------------------------------
// 2) Weight transpose + f32->f16:  in[K][N] -> out[N][K]
__global__ __launch_bounds__(256) void transpose_cvt_kernel(const float* __restrict__ in,
                                                            _Float16* __restrict__ out,
                                                            int K, int N) {
  size_t tid = (size_t)blockIdx.x * 256 + threadIdx.x;
  if (tid >= (size_t)K * N) return;
  int k = (int)(tid % K);
  size_t n = tid / K;
  out[n * K + k] = (_Float16)in[(size_t)k * N + n];
}

// ---------------------------------------------------------------------------
// 3/6) WMMA GEMM: C[M,N] = A[M,K](f16) * Bt[N,K](f16, col-major view) + bias
//      one 16x32 output strip per wave (A fragment shared by two B fragments)
template <bool F16OUT>
__global__ __launch_bounds__(256) void gemm_wmma_kernel(const _Float16* __restrict__ A,
                                                        const _Float16* __restrict__ Bt,
                                                        const float* __restrict__ bias,
                                                        void* __restrict__ Cout,
                                                        int M, int N, int K) {
  const int wid = blockIdx.x * (blockDim.x >> 5) + (threadIdx.x >> 5);
  const int ntn = N >> 5;                 // 32-wide N strips
  const int mt = wid / ntn;
  const int nt = wid % ntn;
  if (mt >= (M >> 4)) return;

  const _Float16* Ab  = A  + (size_t)mt * 16 * K;
  const _Float16* Bb0 = Bt + (size_t)(nt * 32) * K;
  const _Float16* Bb1 = Bt + (size_t)(nt * 32 + 16) * K;

  v8f acc0 = {}, acc1 = {};
  for (int k0 = 0; k0 < K; k0 += 32) {
    v16h a  = load_k32(Ab,  K, k0);
    v16h b0 = load_k32(Bb0, K, k0);
    v16h b1 = load_k32(Bb1, K, k0);
    acc0 = wmma_f16(a, b0, acc0);
    acc1 = wmma_f16(a, b1, acc1);
  }

  const int lane = threadIdx.x & 31;
  const int c = lane & 15, hi = lane >> 4;
  const int colg0 = nt * 32 + c;
  const float bv0 = bias[colg0];
  const float bv1 = bias[colg0 + 16];
#pragma unroll
  for (int i = 0; i < 8; ++i) {
    const size_t rowg = (size_t)mt * 16 + i + 8 * hi;
    const float v0 = acc0[i] + bv0;
    const float v1 = acc1[i] + bv1;
    if (F16OUT) {
      ((_Float16*)Cout)[rowg * N + colg0]      = (_Float16)v0;
      ((_Float16*)Cout)[rowg * N + colg0 + 16] = (_Float16)v1;
    } else {
      ((float*)Cout)[rowg * N + colg0]      = v0;
      ((float*)Cout)[rowg * N + colg0 + 16] = v1;
    }
  }
}

// ---------------------------------------------------------------------------
// 4) RoPE + scatter into attention layouts:
//    Q,K -> [B,H,L,64] f16 (rotated);  V -> [B,H,64,L] f16 (transposed)
__global__ __launch_bounds__(256) void rope_kernel(const _Float16* __restrict__ qkv,
                                                   const float* __restrict__ pos,
                                                   _Float16* __restrict__ Qr,
                                                   _Float16* __restrict__ Kr,
                                                   _Float16* __restrict__ Vt) {
  size_t tid = (size_t)blockIdx.x * 256 + threadIdx.x;
  const size_t total = (size_t)BATCH * SEQ * NHEAD * 32;
  if (tid >= total) return;
  const int j = (int)(tid & 31);        // d in [0,32) (first half)
  size_t t = tid >> 5;
  const int h = (int)(t % NHEAD); t /= NHEAD;
  const int l = (int)(t % SEQ);
  const int b = (int)(t / SEQ);
  const int g = h >> 2;                 // head group -> pos axis (H/3 = 4)

  // fp16 angle, matching the reference's .half() casts
  const _Float16 ph = (_Float16)pos[((size_t)b * SEQ + l) * 3 + g];
  const _Float16 fr = (_Float16)__powf(10000.0f, -(float)j * (1.0f / 32.0f));
  const float ang = (float)(_Float16)(ph * fr);
  const float cs = __cosf(ang), sn = __sinf(ang);

  const size_t base = ((size_t)b * SEQ + l) * NQKV + h * HDIM + j;
  const float q1 = (float)qkv[base],            q2 = (float)qkv[base + 32];
  const float k1 = (float)qkv[base + EMB],      k2 = (float)qkv[base + EMB + 32];
  const float v1 = (float)qkv[base + 2 * EMB],  v2 = (float)qkv[base + 2 * EMB + 32];

  const size_t ob = (((size_t)b * NHEAD + h) * SEQ + l) * HDIM + j;
  Qr[ob]      = (_Float16)(q1 * cs - q2 * sn);
  Qr[ob + 32] = (_Float16)(q2 * cs + q1 * sn);
  Kr[ob]      = (_Float16)(k1 * cs - k2 * sn);
  Kr[ob + 32] = (_Float16)(k2 * cs + k1 * sn);

  const size_t vb = (((size_t)b * NHEAD + h) * HDIM + j) * SEQ + l;
  Vt[vb]                   = (_Float16)v1;
  Vt[vb + (size_t)32 * SEQ] = (_Float16)v2;
}

// ---------------------------------------------------------------------------
// 5) Flash attention: one wave = one (b,h, 32-row q tile); online softmax.
//    32x32 S block via 8 WMMAs / kv-step(32); K/V fragments shared by both
//    16-row blocks. Softmax: lane owns one full S row -> NO cross-lane ops.
//    P relayout C->A through LDS; P*V via 8 WMMAs.
#define FA_WAVES 8
#define QROWS 32
#define SROW 36                         // padded f32 row (bank-conflict free, 16B aligned)
#define PROW 40                         // padded f16 row (80B, 16B aligned)
__global__ __launch_bounds__(FA_WAVES * 32) void flash_kernel(const _Float16* __restrict__ Qr,
                                                              const _Float16* __restrict__ Kr,
                                                              const _Float16* __restrict__ Vt,
                                                              _Float16* __restrict__ attn) {
  __shared__ float    Sbuf[FA_WAVES][QROWS * SROW];   // 4.5 KB / wave
  __shared__ _Float16 Pbuf[FA_WAVES][QROWS * PROW];   // 2.5 KB / wave
  __shared__ float    Abuf[FA_WAVES][QROWS];          // per-row rescale factors

  const int wid  = threadIdx.x >> 5;
  const int lane = threadIdx.x & 31;
  const int gw = blockIdx.x * FA_WAVES + wid;         // 0 .. B*H*(L/32)-1
  const int qt = gw & (SEQ / QROWS - 1);              // 64 q tiles
  const int bh = gw >> 6;

  const _Float16* Qb = Qr + (((size_t)bh * SEQ) + qt * QROWS) * HDIM;
  const _Float16* Kb = Kr + (size_t)bh * SEQ * HDIM;
  const _Float16* Vb = Vt + (size_t)bh * HDIM * SEQ;

  // Q fragments (two 16-row blocks) with 1/sqrt(D)=0.125 folded in
  V16U q00, q01, q10, q11;
  q00.v = load_k32(Qb,              HDIM, 0);
  q01.v = load_k32(Qb,              HDIM, 32);
  q10.v = load_k32(Qb + 16 * HDIM,  HDIM, 0);
  q11.v = load_k32(Qb + 16 * HDIM,  HDIM, 32);
#pragma unroll
  for (int i = 0; i < 16; ++i) {
    q00.e[i] = q00.e[i] * (_Float16)0.125f; q01.e[i] = q01.e[i] * (_Float16)0.125f;
    q10.e[i] = q10.e[i] * (_Float16)0.125f; q11.e[i] = q11.e[i] * (_Float16)0.125f;
  }
  const v16h qa00 = q00.v, qa01 = q01.v, qa10 = q10.v, qa11 = q11.v;

  // row-owner state: lane l owns the full S row l (32 columns)
  float m = -1e30f, sum = 0.f;
  v8f O00 = {}, O01 = {}, O02 = {}, O03 = {};   // rows 0-15,  d 0..63
  v8f O10 = {}, O11 = {}, O12 = {}, O13 = {};   // rows 16-31, d 0..63

  const int c  = lane & 15;              // C-layout column within tile
  const int hi = lane >> 4;
  float*    sl = Sbuf[wid];
  _Float16* pl = Pbuf[wid];
  float*    al = Abuf[wid];

  for (int kv0 = 0; kv0 < SEQ; kv0 += 32) {
    // --- S = q @ k^T (32 x 32): K fragments shared by both row blocks ------
    const _Float16* K0 = Kb + (size_t)kv0 * HDIM;
    const _Float16* K1 = Kb + (size_t)(kv0 + 16) * HDIM;
    v16h kb00 = load_k32(K0, HDIM, 0);
    v16h kb01 = load_k32(K0, HDIM, 32);
    v16h kb10 = load_k32(K1, HDIM, 0);
    v16h kb11 = load_k32(K1, HDIM, 32);
    if (kv0 + 32 < SEQ) {
      __builtin_prefetch(Kb + (size_t)(kv0 + 32) * HDIM, 0, 1);
      __builtin_prefetch(Vb + kv0 + 32, 0, 1);
    }

    v8f S00 = {}, S01 = {}, S10 = {}, S11 = {};
    S00 = wmma_f16(qa00, kb00, S00); S00 = wmma_f16(qa01, kb01, S00);
    S01 = wmma_f16(qa00, kb10, S01); S01 = wmma_f16(qa01, kb11, S01);
    S10 = wmma_f16(qa10, kb00, S10); S10 = wmma_f16(qa11, kb01, S10);
    S11 = wmma_f16(qa10, kb10, S11); S11 = wmma_f16(qa11, kb11, S11);

    // --- C-layout -> LDS f32 ------------------------------------------------
#pragma unroll
    for (int i = 0; i < 8; ++i) {
      const int r0 = i + 8 * hi;
      sl[r0 * SROW + c]             = S00[i];
      sl[r0 * SROW + c + 16]        = S01[i];
      sl[(r0 + 16) * SROW + c]      = S10[i];
      sl[(r0 + 16) * SROW + c + 16] = S11[i];
    }
    asm volatile("s_wait_dscnt 0" ::: "memory");

    // --- row-owner softmax: lane owns full row -> pure in-lane reduction ---
    const float* srow = &sl[lane * SROW];
    float4 f0 = *(const float4*)(srow);
    float4 f1 = *(const float4*)(srow + 4);
    float4 f2 = *(const float4*)(srow + 8);
    float4 f3 = *(const float4*)(srow + 12);
    float4 f4 = *(const float4*)(srow + 16);
    float4 f5 = *(const float4*)(srow + 20);
    float4 f6 = *(const float4*)(srow + 24);
    float4 f7 = *(const float4*)(srow + 28);
    float rmax = fmaxf(fmaxf(fmaxf(f0.x, f0.y), fmaxf(f0.z, f0.w)),
                       fmaxf(fmaxf(f1.x, f1.y), fmaxf(f1.z, f1.w)));
    rmax = fmaxf(rmax, fmaxf(fmaxf(fmaxf(f2.x, f2.y), fmaxf(f2.z, f2.w)),
                             fmaxf(fmaxf(f3.x, f3.y), fmaxf(f3.z, f3.w))));
    rmax = fmaxf(rmax, fmaxf(fmaxf(fmaxf(f4.x, f4.y), fmaxf(f4.z, f4.w)),
                             fmaxf(fmaxf(f5.x, f5.y), fmaxf(f5.z, f5.w))));
    rmax = fmaxf(rmax, fmaxf(fmaxf(fmaxf(f6.x, f6.y), fmaxf(f6.z, f6.w)),
                             fmaxf(fmaxf(f7.x, f7.y), fmaxf(f7.z, f7.w))));

    const float mn = fmaxf(m, rmax);
    const float alpha = __expf(m - mn);
    m = mn;

    float p[32];
    p[0]=__expf(f0.x-mn);  p[1]=__expf(f0.y-mn);  p[2]=__expf(f0.z-mn);  p[3]=__expf(f0.w-mn);
    p[4]=__expf(f1.x-mn);  p[5]=__expf(f1.y-mn);  p[6]=__expf(f1.z-mn);  p[7]=__expf(f1.w-mn);
    p[8]=__expf(f2.x-mn);  p[9]=__expf(f2.y-mn);  p[10]=__expf(f2.z-mn); p[11]=__expf(f2.w-mn);
    p[12]=__expf(f3.x-mn); p[13]=__expf(f3.y-mn); p[14]=__expf(f3.z-mn); p[15]=__expf(f3.w-mn);
    p[16]=__expf(f4.x-mn); p[17]=__expf(f4.y-mn); p[18]=__expf(f4.z-mn); p[19]=__expf(f4.w-mn);
    p[20]=__expf(f5.x-mn); p[21]=__expf(f5.y-mn); p[22]=__expf(f5.z-mn); p[23]=__expf(f5.w-mn);
    p[24]=__expf(f6.x-mn); p[25]=__expf(f6.y-mn); p[26]=__expf(f6.z-mn); p[27]=__expf(f6.w-mn);
    p[28]=__expf(f7.x-mn); p[29]=__expf(f7.y-mn); p[30]=__expf(f7.z-mn); p[31]=__expf(f7.w-mn);
    float rsum = 0.f;
#pragma unroll
    for (int i = 0; i < 32; ++i) rsum += p[i];
    sum = sum * alpha + rsum;

    // --- write P (f16) + alpha back to LDS ---------------------------------
    V8HU pk[4];
#pragma unroll
    for (int q = 0; q < 4; ++q)
#pragma unroll
      for (int i = 0; i < 8; ++i) pk[q].e[i] = (_Float16)p[q * 8 + i];
    _Float16* prow = &pl[lane * PROW];
    *(v8h*)(prow)      = pk[0].v;
    *(v8h*)(prow + 8)  = pk[1].v;
    *(v8h*)(prow + 16) = pk[2].v;
    *(v8h*)(prow + 24) = pk[3].v;
    al[lane] = alpha;
    asm volatile("s_wait_dscnt 0" ::: "memory");

    // --- rescale O accumulators (broadcast alpha per C-layout row) ---------
#pragma unroll
    for (int i = 0; i < 8; ++i) {
      const float a0 = al[i + 8 * hi];
      const float a1 = al[i + 8 * hi + 16];
      O00[i] *= a0; O01[i] *= a0; O02[i] *= a0; O03[i] *= a0;
      O10[i] *= a1; O11[i] *= a1; O12[i] *= a1; O13[i] *= a1;
    }

    // --- P @ V: V fragments shared by both row blocks ----------------------
    v16h pa0 = load_k32(pl,             PROW, 0);
    v16h pa1 = load_k32(pl + 16 * PROW, PROW, 0);
    v16h vb0 = load_k32(Vb + kv0,                      SEQ, 0);
    v16h vb1 = load_k32(Vb + (size_t)16 * SEQ + kv0,   SEQ, 0);
    v16h vb2 = load_k32(Vb + (size_t)32 * SEQ + kv0,   SEQ, 0);
    v16h vb3 = load_k32(Vb + (size_t)48 * SEQ + kv0,   SEQ, 0);

    O00 = wmma_f16(pa0, vb0, O00); O01 = wmma_f16(pa0, vb1, O01);
    O02 = wmma_f16(pa0, vb2, O02); O03 = wmma_f16(pa0, vb3, O03);
    O10 = wmma_f16(pa1, vb0, O10); O11 = wmma_f16(pa1, vb1, O11);
    O12 = wmma_f16(pa1, vb2, O12); O13 = wmma_f16(pa1, vb3, O13);
  }

  // epilogue: broadcast 1/sum per row through LDS, normalize, store f16
  al[lane] = 1.0f / sum;
  asm volatile("s_wait_dscnt 0" ::: "memory");

  const int b_ = bh / NHEAD, h_ = bh % NHEAD;
  _Float16* ob = attn + (((size_t)b_ * SEQ) + qt * QROWS) * EMB + h_ * HDIM;
#pragma unroll
  for (int i = 0; i < 8; ++i) {
    const int r0 = i + 8 * hi;
    const float inv0 = al[r0];
    const float inv1 = al[r0 + 16];
    _Float16* p0 = ob + (size_t)r0 * EMB + c;
    _Float16* p1 = ob + (size_t)(r0 + 16) * EMB + c;
    p0[0]  = (_Float16)(O00[i] * inv0);
    p0[16] = (_Float16)(O01[i] * inv0);
    p0[32] = (_Float16)(O02[i] * inv0);
    p0[48] = (_Float16)(O03[i] * inv0);
    p1[0]  = (_Float16)(O10[i] * inv1);
    p1[16] = (_Float16)(O11[i] * inv1);
    p1[32] = (_Float16)(O12[i] * inv1);
    p1[48] = (_Float16)(O13[i] * inv1);
  }
}

// ---------------------------------------------------------------------------
extern "C" void kernel_launch(void* const* d_in, const int* in_sizes, int n_in,
                              void* d_out, int out_size, void* d_ws, size_t ws_size,
                              hipStream_t stream) {
  const float* feat   = (const float*)d_in[0];   // [4,2048,768]
  const float* pos    = (const float*)d_in[1];   // [4,2048,3]
  const float* ln_g   = (const float*)d_in[2];   // [768]
  const float* ln_b   = (const float*)d_in[3];   // [768]
  const float* w_qkv  = (const float*)d_in[4];   // [768,2304]
  const float* b_qkv  = (const float*)d_in[5];   // [2304]
  const float* w_proj = (const float*)d_in[6];   // [768,768]
  const float* b_proj = (const float*)d_in[7];   // [768]
  float* out = (float*)d_out;                    // [4,2048,768] f32

  // workspace carve-out (256B aligned)
  size_t off = 0;
  auto alloc = [&](size_t bytes) -> char* {
    char* p = (char*)d_ws + off;
    off += (bytes + 255) & ~(size_t)255;
    return p;
  };
  _Float16* xln    = (_Float16*)alloc((size_t)MROWS * EMB * 2);
  _Float16* wqkvT  = (_Float16*)alloc((size_t)NQKV * EMB * 2);
  _Float16* wprojT = (_Float16*)alloc((size_t)EMB * EMB * 2);
  _Float16* qkv    = (_Float16*)alloc((size_t)MROWS * NQKV * 2);
  _Float16* Qr     = (_Float16*)alloc((size_t)BATCH * NHEAD * SEQ * HDIM * 2);
  _Float16* Kr     = (_Float16*)alloc((size_t)BATCH * NHEAD * SEQ * HDIM * 2);
  _Float16* Vt     = (_Float16*)alloc((size_t)BATCH * NHEAD * SEQ * HDIM * 2);
  _Float16* attn   = (_Float16*)alloc((size_t)MROWS * EMB * 2);
  (void)ws_size; (void)in_sizes; (void)n_in; (void)out_size;

  // 1) LayerNorm
  ln_kernel<<<MROWS, 256, 0, stream>>>(feat, ln_g, ln_b, xln);

  // 2) weight transposes (f32 -> f16, [K,N] -> [N,K])
  {
    size_t n1 = (size_t)EMB * NQKV;
    transpose_cvt_kernel<<<(unsigned)((n1 + 255) / 256), 256, 0, stream>>>(w_qkv, wqkvT, EMB, NQKV);
    size_t n2 = (size_t)EMB * EMB;
    transpose_cvt_kernel<<<(unsigned)((n2 + 255) / 256), 256, 0, stream>>>(w_proj, wprojT, EMB, EMB);
  }

  // 3) QKV GEMM (8192 x 2304 x 768), f16 out
  {
    const int tiles = (MROWS / 16) * (NQKV / 32);   // 36864 waves
    gemm_wmma_kernel<true><<<tiles / 8, 256, 0, stream>>>(xln, wqkvT, b_qkv, (void*)qkv,
                                                          MROWS, NQKV, EMB);
  }

  // 4) RoPE + layout scatter
  {
    const size_t total = (size_t)BATCH * SEQ * NHEAD * 32;
    rope_kernel<<<(unsigned)((total + 255) / 256), 256, 0, stream>>>(qkv, pos, Qr, Kr, Vt);
  }

  // 5) flash attention (3072 waves, FA_WAVES per block)
  flash_kernel<<<(BATCH * NHEAD * (SEQ / QROWS)) / FA_WAVES, FA_WAVES * 32, 0, stream>>>(Qr, Kr, Vt, attn);

  // 6) output projection (8192 x 768 x 768), f32 out + bias
  {
    const int tiles = (MROWS / 16) * (EMB / 32);    // 12288 waves
    gemm_wmma_kernel<false><<<tiles / 8, 256, 0, stream>>>(attn, wprojT, b_proj, (void*)out,
                                                           MROWS, EMB, EMB);
  }
}
